// LinearCrossEntropy_38929583571195
// MI455X (gfx1250) — matrix operand order repaired and checked by reading, not compiled, not run on hardware
//
#include <hip/hip_runtime.h>

// ---------------------------------------------------------------------------
// Fused linear cross-entropy (CCE) for gfx1250:
//   logits = e @ c^T  (bf16 WMMA, f32 accumulate), online logsumexp over V,
//   target-logit capture, mean NLL.
// Round 3: global_prefetch of next c-chunk overlaps staging latency with
//          compute; 16B (b128) LDS staging stores.
// ---------------------------------------------------------------------------

typedef __attribute__((ext_vector_type(16))) __bf16 v16bf;
typedef __attribute__((ext_vector_type(8)))  __bf16 v8bf;
typedef __attribute__((ext_vector_type(8)))  float  v8f;

#define N_TOK   8192
#define DDIM    1024
#define VDIM    50257
#define NSPLIT  4
#define VC      12608          // per-split V range, multiple of 64 (4*12608 >= 50257)
#define NIT     (VC / 64)      // 197 iterations, uniform across splits
#define BROWS   64             // e rows per block
#define BCOLS   64             // c rows (logit cols) per iteration
#define ASTR    1032           // padded LDS row stride (bf16 elems), 16B-friendly
#define BSTR    1032
#define IGNORE_INDEX (-100)

__device__ __forceinline__ unsigned short f2bf_bits(float x) {
    union { float f; unsigned int u; } v; v.f = x;
    unsigned int r = v.u + 0x7FFFu + ((v.u >> 16) & 1u);   // round-to-nearest-even
    return (unsigned short)(r >> 16);
}

__device__ __forceinline__ unsigned int pack2bf(float a, float b) {
    return (unsigned int)f2bf_bits(a) | ((unsigned int)f2bf_bits(b) << 16);
}

__device__ __forceinline__ v16bf ld_fragA(const __bf16* base, int kc) {
    // A layout: lane half h holds K in {8h..8h+7} U {16+8h..23+8h} (base pre-offset by 8h)
    v8bf lo = *(const v8bf*)(base + kc * 32);
    v8bf hi = *(const v8bf*)(base + kc * 32 + 16);
    return __builtin_shufflevector(lo, hi, 0,1,2,3,4,5,6,7,8,9,10,11,12,13,14,15);
}

__device__ __forceinline__ v16bf ld_fragB(const __bf16* base, int kc) {
    // B layout: lane half h holds K in {8h..8h+7} U {16+8h..23+8h} (base pre-offset by 8h)
    v8bf lo = *(const v8bf*)(base + kc * 32);
    v8bf hi = *(const v8bf*)(base + kc * 32 + 16);
    return __builtin_shufflevector(lo, hi, 0,1,2,3,4,5,6,7,8,9,10,11,12,13,14,15);
}

// ---------------------------------------------------------------------------
// Kernel 1: main GEMM + online softmax partials.
// grid = (N/BROWS, NSPLIT), block = 256 (8 waves).
// Wave w: rowgroup = w&1 (32 rows), colgroup = w>>1 (16 cols).
// ws layout: partials[split][row][3] = {running_max, running_sumexp, tgt_logit}
// ---------------------------------------------------------------------------
__global__ __launch_bounds__(256)
void cce_main_kernel(const float* __restrict__ e,
                     const float* __restrict__ c,
                     const int*   __restrict__ targets,
                     float*       __restrict__ ws)
{
    extern __shared__ __align__(16) char smem_raw[];
    __bf16* As = (__bf16*)smem_raw;            // BROWS x ASTR bf16
    __bf16* Bs = As + BROWS * ASTR;            // BCOLS x BSTR bf16

    const int tid      = threadIdx.x;
    const int lane     = tid & 31;
    const int wave     = tid >> 5;             // 0..7
    const int rowgroup = wave & 1;             // 0..1  (32-row groups)
    const int colgroup = wave >> 1;            // 0..3  (16-col groups)
    const int h        = lane >> 4;            // half-wave
    const int ln       = lane & 15;

    const int rb    = blockIdx.x;              // row block
    const int split = blockIdx.y;
    const int c0    = split * VC;
    const int cend  = min(VDIM, c0 + VC);

    // ---- stage A tile (e rows) as bf16, once (8 floats -> one 16B store) ----
    {
        const float4* eg = (const float4*)(e + (size_t)rb * BROWS * DDIM);
        for (int i = tid; i < BROWS * DDIM / 8; i += 256) {
            float4 v0 = eg[2 * i];
            float4 v1 = eg[2 * i + 1];
            int r = i >> 7;                    // / (1024/8)
            int k = (i & 127) * 8;
            uint4 p;
            p.x = pack2bf(v0.x, v0.y);
            p.y = pack2bf(v0.z, v0.w);
            p.z = pack2bf(v1.x, v1.y);
            p.w = pack2bf(v1.z, v1.w);
            *(uint4*)(As + r * ASTR + k) = p;
        }
    }

    // ---- per-lane online-softmax state: 16 rows (2 tiles x 8) x 1 col slot ----
    int   tg[16];
    float m[16], s[16], tacc[16];
#pragma unroll
    for (int t = 0; t < 2; ++t) {
#pragma unroll
        for (int i = 0; i < 8; ++i) {
            int row = rb * BROWS + rowgroup * 32 + t * 16 + 8 * h + i;
            tg[t * 8 + i]   = targets[row];
            m[t * 8 + i]    = -3.0e38f;
            s[t * 8 + i]    = 0.0f;
            tacc[t * 8 + i] = 0.0f;
        }
    }

    const int colslot = c0 + colgroup * 16 + ln;   // this lane's column (+64*it)

    const __bf16* Ab0 = As + (rowgroup * 32 + ln) * ASTR + 8 * h;        // tile 0 rows
    const __bf16* Ab1 = Ab0 + 16 * ASTR;                                  // tile 1 rows
    const __bf16* Bb  = Bs + (colgroup * 16 + ln) * BSTR + 8 * h;

    for (int it = 0; it < NIT; ++it) {
        __syncthreads();                       // previous compute done before B overwrite
        // ---- stage B chunk: 64 rows of c as bf16 (8 floats -> one 16B store) ----
        {
            const int base_c = c0 + it * 64;
            for (int i = tid; i < BCOLS * DDIM / 8; i += 256) {
                int r  = i >> 7;
                int kv = (i & 127);
                int crow = base_c + r;
                float4 v0, v1;
                if (crow < VDIM) {
                    const float4* cp = (const float4*)(c + (size_t)crow * DDIM);
                    v0 = cp[2 * kv];
                    v1 = cp[2 * kv + 1];
                } else {
                    v0.x = v0.y = v0.z = v0.w = 0.0f;
                    v1 = v0;
                }
                uint4 p;
                p.x = pack2bf(v0.x, v0.y);
                p.y = pack2bf(v0.z, v0.w);
                p.z = pack2bf(v1.x, v1.y);
                p.w = pack2bf(v1.z, v1.w);
                *(uint4*)(Bs + r * BSTR + kv * 8) = p;
            }
        }

        // ---- prefetch next iteration's c-chunk into cache (overlaps compute) ----
        if (it + 1 < NIT) {
            const int base_n = c0 + (it + 1) * 64;
            const char* cbytes = (const char*)c;
#pragma unroll
            for (int j = 0; j < 8; ++j) {
                int lid  = tid + 256 * j;              // 2048 lines of 128B = 256KB
                int r    = lid >> 5;                   // 32 lines per 4KB row
                int off  = (lid & 31) * 128;
                int crow = base_n + r;
                crow = (crow < VDIM) ? crow : (VDIM - 1);   // stay in-bounds
                __builtin_prefetch(cbytes + (size_t)crow * DDIM * 4 + off, 0, 3);
            }
        }
        __syncthreads();

        // ---- 32x16 logits (2 tiles) over K=1024, software-pipelined ----
        v8f acc0 = {0.f,0.f,0.f,0.f,0.f,0.f,0.f,0.f};
        v8f acc1 = {0.f,0.f,0.f,0.f,0.f,0.f,0.f,0.f};
        v16bf a0 = ld_fragA(Ab0, 0);
        v16bf a1 = ld_fragA(Ab1, 0);
        v16bf b0 = ld_fragB(Bb,  0);
#pragma unroll
        for (int kc = 0; kc < 31; ++kc) {
            v16bf an0 = ld_fragA(Ab0, kc + 1);
            v16bf an1 = ld_fragA(Ab1, kc + 1);
            v16bf bn  = ld_fragB(Bb,  kc + 1);
            acc0 = __builtin_amdgcn_wmma_f32_16x16x32_bf16(
                false, a0, false, b0, (short)0, acc0, false, false);
            acc1 = __builtin_amdgcn_wmma_f32_16x16x32_bf16(
                false, a1, false, b0, (short)0, acc1, false, false);
            a0 = an0; a1 = an1; b0 = bn;
        }
        acc0 = __builtin_amdgcn_wmma_f32_16x16x32_bf16(
            false, a0, false, b0, (short)0, acc0, false, false);
        acc1 = __builtin_amdgcn_wmma_f32_16x16x32_bf16(
            false, a1, false, b0, (short)0, acc1, false, false);

        // ---- online softmax update + target capture ----
        int col = colslot + it * 64;
        if (col < cend) {
#pragma unroll
            for (int i = 0; i < 8; ++i) {
                float x  = acc0[i];
                float mn = fmaxf(m[i], x);
                s[i] = s[i] * __expf(m[i] - mn) + __expf(x - mn);
                m[i] = mn;
                if (col == tg[i]) tacc[i] += x;
            }
#pragma unroll
            for (int i = 0; i < 8; ++i) {
                float x  = acc1[i];
                float mn = fmaxf(m[8 + i], x);
                s[8 + i] = s[8 + i] * __expf(m[8 + i] - mn) + __expf(x - mn);
                m[8 + i] = mn;
                if (col == tg[8 + i]) tacc[8 + i] += x;
            }
        }
    }
    __syncthreads();   // all waves done computing before LDS reuse

    // ---- reduce across the 16 lanes that share each row (within half-wave) ----
#pragma unroll
    for (int i = 0; i < 16; ++i) {
#pragma unroll
        for (int off = 1; off < 16; off <<= 1) {
            float mo = __shfl_xor(m[i],    off, 32);
            float so = __shfl_xor(s[i],    off, 32);
            float to = __shfl_xor(tacc[i], off, 32);
            float mn = fmaxf(m[i], mo);
            s[i] = s[i] * __expf(m[i] - mn) + so * __expf(mo - mn);
            m[i] = mn;
            tacc[i] += to;
        }
    }

    // ---- merge the four col-group waves per row via LDS, emit per-split partials ----
    float* mb = (float*)smem_raw;       // [4][64]
    float* sb = mb + 4 * 64;
    float* tb = sb + 4 * 64;
    if (ln == 0) {
#pragma unroll
        for (int t = 0; t < 2; ++t) {
#pragma unroll
            for (int i = 0; i < 8; ++i) {
                int rl = rowgroup * 32 + t * 16 + 8 * h + i;
                mb[colgroup * 64 + rl] = m[t * 8 + i];
                sb[colgroup * 64 + rl] = s[t * 8 + i];
                tb[colgroup * 64 + rl] = tacc[t * 8 + i];
            }
        }
    }
    __syncthreads();
    if (tid < 64) {
        float M = -3.0e38f, S = 0.0f, T = 0.0f;
#pragma unroll
        for (int g = 0; g < 4; ++g) {
            float mm = mb[g * 64 + tid];
            float ss = sb[g * 64 + tid];
            float Mn = fmaxf(M, mm);
            S = S * __expf(M - Mn) + ss * __expf(mm - Mn);
            M = Mn;
            T += tb[g * 64 + tid];
        }
        int row  = rb * BROWS + tid;
        float* o = ws + ((size_t)split * N_TOK + row) * 3;
        o[0] = M; o[1] = S; o[2] = T;
    }
}

// ---------------------------------------------------------------------------
// Kernel 2: merge splits per row, compute NLL, reduce to (sum, count).
// ---------------------------------------------------------------------------
__global__ __launch_bounds__(256)
void cce_reduce_kernel(const float* __restrict__ ws,
                       const int*   __restrict__ targets,
                       float*       __restrict__ acc)
{
    int r = blockIdx.x * 256 + threadIdx.x;
    float nll = 0.0f, vld = 0.0f;
    if (r < N_TOK) {
        float M = -3.0e38f, S = 0.0f, T = 0.0f;
#pragma unroll
        for (int sp = 0; sp < NSPLIT; ++sp) {
            const float* p = ws + ((size_t)sp * N_TOK + r) * 3;
            float mm = p[0], ss = p[1], tt = p[2];
            float Mn = fmaxf(M, mm);
            S = S * __expf(M - Mn) + ss * __expf(mm - Mn);
            M = Mn;
            T += tt;
        }
        float lse = M + __logf(S);
        bool valid = (targets[r] != IGNORE_INDEX);
        nll = valid ? (lse - T) : 0.0f;
        vld = valid ? 1.0f : 0.0f;
    }
    __shared__ float red0[256];
    __shared__ float red1[256];
    red0[threadIdx.x] = nll;
    red1[threadIdx.x] = vld;
    __syncthreads();
    for (int st = 128; st > 0; st >>= 1) {
        if ((int)threadIdx.x < st) {
            red0[threadIdx.x] += red0[threadIdx.x + st];
            red1[threadIdx.x] += red1[threadIdx.x + st];
        }
        __syncthreads();
    }
    if (threadIdx.x == 0) {
        atomicAdd(&acc[0], red0[0]);
        atomicAdd(&acc[1], red1[0]);
    }
}

__global__ void cce_zero_kernel(float* __restrict__ acc) {
    acc[0] = 0.0f;
    acc[1] = 0.0f;
}

__global__ void cce_final_kernel(const float* __restrict__ acc,
                                 float* __restrict__ out) {
    out[0] = acc[0] / fmaxf(acc[1], 1.0f);
}

// ---------------------------------------------------------------------------
extern "C" void kernel_launch(void* const* d_in, const int* in_sizes, int n_in,
                              void* d_out, int out_size, void* d_ws, size_t ws_size,
                              hipStream_t stream) {
    const float* e  = (const float*)d_in[0];
    const float* c  = (const float*)d_in[1];
    const int*   tg = (const int*)d_in[2];
    float* out = (float*)d_out;
    float* ws  = (float*)d_ws;                         // partials: NSPLIT*N_TOK*3 floats
    float* acc = ws + (size_t)NSPLIT * N_TOK * 3;      // + 2 accumulator floats

    const size_t ldsBytes = (size_t)(BROWS * ASTR + BCOLS * BSTR) * 2;   // 264192 B

    cce_zero_kernel<<<1, 1, 0, stream>>>(acc);
    cce_main_kernel<<<dim3(N_TOK / BROWS, NSPLIT), 256, ldsBytes, stream>>>(e, c, tg, ws);
    cce_reduce_kernel<<<N_TOK / 256, 256, 0, stream>>>(ws, tg, acc);
    cce_final_kernel<<<1, 1, 0, stream>>>(acc, out);
}
